// KNN_itc_43121471652316
// MI455X (gfx1250) — matrix operand order repaired and checked by reading, not compiled, not run on hardware
//
#include <hip/hip_runtime.h>
#include <hip/hip_bf16.h>

// ---------------------------------------------------------------------------
// KNN image-to-class similarity (cosine GEMM + top-3 reduction) for gfx1250.
//   out[b][n] = sum_m topk3_q( qd[b] (441x640) . sd[n] (441x640)^T )[.,m]
// Compute-bound (~1000 FLOP/byte) -> V_WMMA_F32_16X16X32_F16 matrix pipe.
// K-outer / row-tile-inner nest (nothing loop-invariant -> no LICM spills),
// 2 column tiles x 7 row tiles per wave: each A fragment feeds 2 WMMAs,
// each B fragment feeds 7 WMMAs  (~1.3 b128 loads per WMMA).
// ---------------------------------------------------------------------------

typedef __attribute__((ext_vector_type(16))) _Float16 v16h;
typedef __attribute__((ext_vector_type(8)))  _Float16 v8h;
typedef __attribute__((ext_vector_type(8)))  float    v8f;

#define KNN_C    640
#define KNN_HW   441
#define KNN_MP   448   // HW padded to multiple of 16
#define KNN_B    75
#define KNN_N    5

// ---------------------------------------------------------------------------
// Kernel 1: per-descriptor L2 normalization + f16 conversion + transpose.
// in : [nimg, C, HW] fp32   out: [nimg, MP, C] f16 (rows >= HW zero-padded)
// grid.x = nimg * MP, block = 128
// ---------------------------------------------------------------------------
__global__ void knn_normalize_kernel(const float* __restrict__ in,
                                     _Float16* __restrict__ out) {
    constexpr int NT  = 128;
    constexpr int PER = KNN_C / NT;       // 5 channels per thread
    const int b   = blockIdx.x / KNN_MP;
    const int p   = blockIdx.x % KNN_MP;
    const int tid = threadIdx.x;
    _Float16* orow = out + (size_t)blockIdx.x * KNN_C;

    if (p >= KNN_HW) {                    // uniform across the block
        for (int i = tid; i < KNN_C; i += NT) orow[i] = (_Float16)0.0f;
        return;
    }

    const float* base = in + ((size_t)b * KNN_C) * KNN_HW + p;
    float vals[PER];
    float s = 0.0f;
#pragma unroll
    for (int i = 0; i < PER; ++i) {
        float v = base[(size_t)(tid + i * NT) * KNN_HW];
        vals[i] = v;
        s += v * v;
    }

    __shared__ float red[NT];
    red[tid] = s;
    __syncthreads();
#pragma unroll
    for (int off = NT / 2; off > 0; off >>= 1) {
        if (tid < off) red[tid] += red[tid + off];
        __syncthreads();
    }
    const float rn = rsqrtf(red[0]);
#pragma unroll
    for (int i = 0; i < PER; ++i)
        orow[tid + i * NT] = (_Float16)(vals[i] * rn);
}

// ---------------------------------------------------------------------------
// Fragment loaders matching the CDNA5 16-bit WMMA VGPR layouts (wave32).
// A (16x32 MxK): lanes 0-15 hold K 0-7 (V0-3) & 16-23 (V4-7); lanes 16-31
//                hold K 8-15 & 24-31.
// B (32x16 KxN): same K split as A (lanes 0-15: K 0-7 & 16-23; 16-31: 8-15 & 24-31).
// rowp = start of the (row-major, C-contiguous) descriptor row; kb = K base.
// ---------------------------------------------------------------------------
__device__ __forceinline__ v16h knn_load_frag_a(const _Float16* rowp, int kb, int h) {
    v8h lo = *(const v8h*)(rowp + kb + h * 8);        // K: kb+h*8    .. +8
    v8h hi = *(const v8h*)(rowp + kb + 16 + h * 8);   // K: kb+16+h*8 .. +8
    v16h f;
#pragma unroll
    for (int i = 0; i < 8; ++i) { f[i] = lo[i]; f[8 + i] = hi[i]; }
    return f;
}

__device__ __forceinline__ v16h knn_load_frag_b(const _Float16* rowp, int kb, int h) {
    v8h lo = *(const v8h*)(rowp + kb + h * 8);        // K: kb+h*8    .. +8
    v8h hi = *(const v8h*)(rowp + kb + 16 + h * 8);   // K: kb+16+h*8 .. +8
    v16h f;
#pragma unroll
    for (int i = 0; i < 8; ++i) { f[i] = lo[i]; f[8 + i] = hi[i]; }
    return f;
}

// Branchless sorted-insert of x into running top-3 (t0 >= t1 >= t2).
__device__ __forceinline__ void knn_top3_insert(float x, float& t0, float& t1, float& t2) {
    float m0  = fmaxf(x, t0);
    float mn0 = fminf(x, t0);
    float m1  = fmaxf(mn0, t1);
    float mn1 = fminf(mn0, t1);
    float m2  = fmaxf(mn1, t2);
    t0 = m0; t1 = m1; t2 = m2;
}

// ---------------------------------------------------------------------------
// Kernel 2: fused cosine GEMM (f16 WMMA, f32 acc) + top-3-over-q + sum-over-m.
// grid.x = B*N (375), block = 128 (4 waves).
// Wave w owns row tiles rt = 7w .. 7w+6; block loops over column-tile PAIRS.
// Per K-step: 2 B fragments + 7 A fragments -> 14 WMMAs.
// D-matrix layout: lane L -> column n = L&15; VGPR v -> row (L>>4)*8 + v.
// ---------------------------------------------------------------------------
__global__ void __launch_bounds__(128, 1)
knn_gemm_topk_kernel(const _Float16* __restrict__ qd,
                     const _Float16* __restrict__ sup,
                     float* __restrict__ out) {
    constexpr int KT  = KNN_C / 32;       // 20 K-steps
    constexpr int CT  = KNN_MP / 16;      // 28 column tiles (m axis)
    constexpr int RPW = 7;                // row tiles per wave
    const float NEG = -__builtin_inff();

    const int blk = blockIdx.x;           // 0..374
    const int b   = blk / KNN_N;
    const int n   = blk % KNN_N;
    const _Float16* A  = qd  + (size_t)b * KNN_MP * KNN_C;
    const _Float16* Bm = sup + (size_t)n * KNN_MP * KNN_C;

    const int tid  = threadIdx.x;
    const int wave = tid >> 5;            // 0..3
    const int lane = tid & 31;
    const int nn   = lane & 15;           // column / A-row selector
    const int h    = lane >> 4;           // half selector

    // Per-wave A row pointers (row tiles 7w .. 7w+6, row = rt*16 + nn).
    const _Float16* arowp[RPW];
#pragma unroll
    for (int r = 0; r < RPW; ++r)
        arowp[r] = A + (size_t)((wave * RPW + r) * 16 + nn) * KNN_C;

    __shared__ float mrg[4][2][16][3];    // [wave][col-tile][column][triple]
    float total = 0.0f;                   // meaningful on tid 0..31

    for (int ct = 0; ct < CT; ct += 2) {
        const _Float16* brow0 = Bm + (size_t)(ct * 16 + nn) * KNN_C;
        const _Float16* brow1 = brow0 + 16 * KNN_C;
        if (ct + 2 < CT)
            __builtin_prefetch(brow0 + 32 * KNN_C, 0, 1);  // next col-tile pair

        v8f acc[2][RPW];
        const v8f vzero = {};
#pragma unroll
        for (int c = 0; c < 2; ++c)
#pragma unroll
            for (int r = 0; r < RPW; ++r) acc[c][r] = vzero;

        // K outer, row-tile inner: every load depends on the loop indices,
        // so nothing is invariant (no LICM spills). 2 B frags -> 14 WMMAs.
#pragma unroll 2
        for (int ks = 0; ks < KT; ++ks) {
            const int kb = ks * 32;
            v16h bf0 = knn_load_frag_b(brow0, kb, h);
            v16h bf1 = knn_load_frag_b(brow1, kb, h);
#pragma unroll
            for (int r = 0; r < RPW; ++r) {
                v16h af = knn_load_frag_a(arowp[r], kb, h);
                acc[0][r] = __builtin_amdgcn_wmma_f32_16x16x32_f16(
                                false, af, false, bf0, (short)0, acc[0][r],
                                false, false);
                acc[1][r] = __builtin_amdgcn_wmma_f32_16x16x32_f16(
                                false, af, false, bf1, (short)0, acc[1][r],
                                false, false);
            }
        }

        // Per-lane running top-3 over this wave's 112 rows, per column tile.
#pragma unroll
        for (int c = 0; c < 2; ++c) {
            float t0 = NEG, t1 = NEG, t2 = NEG;
#pragma unroll
            for (int r = 0; r < RPW; ++r) {
                const int rb = (wave * RPW + r) * 16 + h * 8;
#pragma unroll
                for (int v = 0; v < 8; ++v) {
                    float x = (rb + v < KNN_HW) ? acc[c][r][v] : NEG;  // pad rows
                    knn_top3_insert(x, t0, t1, t2);
                }
            }
            // Lanes L and L^16 hold the same column over disjoint row halves.
            float o0 = __shfl_xor(t0, 16, 32);
            float o1 = __shfl_xor(t1, 16, 32);
            float o2 = __shfl_xor(t2, 16, 32);
            knn_top3_insert(o0, t0, t1, t2);
            knn_top3_insert(o1, t0, t1, t2);
            knn_top3_insert(o2, t0, t1, t2);

            if (h == 0) {
                mrg[wave][c][nn][0] = t0;
                mrg[wave][c][nn][1] = t1;
                mrg[wave][c][nn][2] = t2;
            }
        }
        __syncthreads();

        if (tid < 32) {                   // wave 0: lanes 0-15 -> ct, 16-31 -> ct+1
            const int c  = tid >> 4;
            const int cn = tid & 15;
            float s0 = NEG, s1 = NEG, s2 = NEG;
#pragma unroll
            for (int w2 = 0; w2 < 4; ++w2) {
                knn_top3_insert(mrg[w2][c][cn][0], s0, s1, s2);
                knn_top3_insert(mrg[w2][c][cn][1], s0, s1, s2);
                knn_top3_insert(mrg[w2][c][cn][2], s0, s1, s2);
            }
            if ((ct + c) * 16 + cn < KNN_HW)   // mask pad columns
                total += s0 + s1 + s2;
        }
        __syncthreads();                  // protect mrg[] before next writes
    }

    // Reduce the 32 per-column totals held by wave 0.
    total += __shfl_down(total, 16, 32);
    total += __shfl_down(total,  8, 32);
    total += __shfl_down(total,  4, 32);
    total += __shfl_down(total,  2, 32);
    total += __shfl_down(total,  1, 32);
    if (tid == 0) out[b * KNN_N + n] = total;   // exp(mean(log(x))) == x, qAV=1
}

// ---------------------------------------------------------------------------
extern "C" void kernel_launch(void* const* d_in, const int* in_sizes, int n_in,
                              void* d_out, int out_size, void* d_ws, size_t ws_size,
                              hipStream_t stream) {
    const float* q = (const float*)d_in[0];   // [75, 640, 21, 21]
    const float* S = (const float*)d_in[1];   // [5, 640, 21, 21]
    float* out = (float*)d_out;               // [75, 5]

    _Float16* qd  = (_Float16*)d_ws;                          // [75][448][640]
    _Float16* sup = qd + (size_t)KNN_B * KNN_MP * KNN_C;      // [5][448][640]

    knn_normalize_kernel<<<KNN_B * KNN_MP, 128, 0, stream>>>(q, qd);
    knn_normalize_kernel<<<KNN_N * KNN_MP, 128, 0, stream>>>(S, sup);
    knn_gemm_topk_kernel<<<KNN_B * KNN_N, 128, 0, stream>>>(qd, sup, out);
}
